// MultiHeadSelfAttention_35356170781281
// MI455X (gfx1250) — matrix operand order repaired and checked by reading, not compiled, not run on hardware
//
#include <hip/hip_runtime.h>
#include <hip/hip_bf16.h>

#define B_ 16
#define N_ 1024
#define D_ 768
#define H_ 12
#define HD_ 64
#define SCALE_ 0.125f

#define GK 40  // GEMM LDS k-stride (bf16): 80 B row = 16B aligned, bank-conflict free
#define AK 72  // attention LDS k-stride (bf16): 144 B row = 16B aligned, conflict free

typedef __attribute__((ext_vector_type(16))) __bf16 v16bf;
typedef __attribute__((ext_vector_type(8))) float v8f;
typedef int v4i_vs __attribute__((__vector_size__(16)));  // matches builtin param type

#if __has_builtin(__builtin_amdgcn_global_load_async_to_lds_b128)
#define HAVE_ASYNC 1
#else
#define HAVE_ASYNC 0
#endif

// ---- WMMA fragment loaders (CDNA5 wave32 layouts, CDNA5 ISA) ----
// All call sites keep 16-byte alignment so these lower to 2x ds_load_b128.

// A 16x32 bf16: lane%16 = row, element j -> k = (j&7) + 16*(j>>3) + 8*(lane>>4)
__device__ __forceinline__ v16bf frag_a(const __bf16* base, int stride, int kOff) {
  const int lane = threadIdx.x & 31;
  const __bf16* p = base + (lane & 15) * stride + kOff + ((lane >> 4) << 3);
  union { uint4 u[2]; v16bf v; } u;
  u.u[0] = *(const uint4*)p;         // j = 0..7
  u.u[1] = *(const uint4*)(p + 16);  // j = 8..15
  return u.v;
}

// B 32x16 bf16 (LDS stored [n][k]): lane%16 = col n, element j -> k = (j&7) + 16*(j>>3) + 8*(lane>>4)
__device__ __forceinline__ v16bf frag_b(const __bf16* base, int stride, int kOff) {
  const int lane = threadIdx.x & 31;
  const __bf16* p = base + (lane & 15) * stride + kOff + ((lane >> 4) << 3);
  union { uint4 u[2]; v16bf v; } u;
  u.u[0] = *(const uint4*)p;
  u.u[1] = *(const uint4*)(p + 16);
  return u.v;
}

__device__ __forceinline__ v8f wmma_bf16(v16bf a, v16bf b, v8f c) {
  return __builtin_amdgcn_wmma_f32_16x16x32_bf16(false, a, false, b, (short)0, c,
                                                 false, false);
}

// ---------------------------------------------------------------------------
// Kernel 1: QKV GEMM.  C[16384,2304] = X[16384,768] @ W[768,2304] + b
// Q,K -> [B,H,N,64] bf16 (Q pre-scaled by 1/8);  V -> [B,H,64,N] bf16 (transposed)
// ---------------------------------------------------------------------------
__global__ __launch_bounds__(256) void qkv_kernel(
    const float* __restrict__ X, const float* __restrict__ W,
    const float* __restrict__ bias, __bf16* __restrict__ Qw,
    __bf16* __restrict__ Kw, __bf16* __restrict__ Vw) {
  __shared__ __bf16 As[128 * GK];
  __shared__ __bf16 Bs[128 * GK];
  const int tid = threadIdx.x;
  const int lane = tid & 31;
  const int w = tid >> 5;
  const int bm = blockIdx.y * 128;
  const int bn = blockIdx.x * 128;
  const int waveM = (w >> 1) * 32;
  const int waveN = (w & 1) * 64;
  const int NC = 3 * D_;

  const int ar = tid >> 3, ac4 = (tid & 7) * 4;    // A stage: float4 per thread
  const int bk = tid >> 5, bn4 = (tid & 31) * 4;   // B stage: float4 per thread

  v8f acc[2][4] = {};

  for (int k0 = 0; k0 < D_; k0 += 32) {
    __syncthreads();
#pragma unroll
    for (int p = 0; p < 4; ++p) {
      int r = ar + 32 * p;
      const float4 v = *(const float4*)&X[(long)(bm + r) * D_ + k0 + ac4];
      __bf16* d = &As[r * GK + ac4];
      d[0] = (__bf16)v.x; d[1] = (__bf16)v.y; d[2] = (__bf16)v.z; d[3] = (__bf16)v.w;
    }
#pragma unroll
    for (int p = 0; p < 4; ++p) {
      int k = bk + 8 * p;
      const float4 v = *(const float4*)&W[(long)(k0 + k) * NC + bn + bn4];
      Bs[(bn4 + 0) * GK + k] = (__bf16)v.x;
      Bs[(bn4 + 1) * GK + k] = (__bf16)v.y;
      Bs[(bn4 + 2) * GK + k] = (__bf16)v.z;
      Bs[(bn4 + 3) * GK + k] = (__bf16)v.w;
    }
    if (k0 + 32 < D_) {
      __builtin_prefetch(&X[(long)(bm + ar) * D_ + k0 + 32 + ac4]);
      __builtin_prefetch(&W[(long)(k0 + 32 + bk) * NC + bn + bn4]);
    }
    __syncthreads();
    v16bf a0 = frag_a(As + (waveM + 0) * GK, GK, 0);
    v16bf a1 = frag_a(As + (waveM + 16) * GK, GK, 0);
#pragma unroll
    for (int j = 0; j < 4; ++j) {
      v16bf b = frag_b(Bs + (waveN + 16 * j) * GK, GK, 0);
      acc[0][j] = wmma_bf16(a0, b, acc[0][j]);
      acc[1][j] = wmma_bf16(a1, b, acc[1][j]);
    }
  }

  const int half = lane >> 4, ln = lane & 15;
#pragma unroll
  for (int i = 0; i < 2; ++i)
#pragma unroll
    for (int j = 0; j < 4; ++j)
#pragma unroll
      for (int r = 0; r < 8; ++r) {
        int m = bm + waveM + 16 * i + r + 8 * half;
        int c = bn + waveN + 16 * j + ln;
        float v = acc[i][j][r] + bias[c];
        int s = c / D_;
        int within = c - s * D_;
        int h = within >> 6, d = within & 63;
        int b = m >> 10, nq = m & 1023;
        if (s == 2) {
          // V stored transposed: [B,H,Hd,N]
          Vw[((long)(b * H_ + h) * HD_ + d) * N_ + nq] = (__bf16)v;
        } else {
          long dst = ((long)(b * H_ + h) * N_ + nq) * HD_ + d;
          if (s == 0) { Qw[dst] = (__bf16)(v * SCALE_); } else { Kw[dst] = (__bf16)v; }
        }
      }
}

// ---------------------------------------------------------------------------
// Async tile staging: 64 rows x 64 bf16, row-contiguous, into LDS stride AK.
// 128 threads x 4 chunks of 16 B.  ASYNCcnt-tracked on CDNA5.
// ---------------------------------------------------------------------------
__device__ __forceinline__ void stage_tile(const __bf16* __restrict__ g, int gstride,
                                           __bf16* l, int tid) {
#pragma unroll
  for (int p = 0; p < 4; ++p) {
    int i = tid + p * 128;
    int r = i >> 3, d0 = (i & 7) * 8;
#if HAVE_ASYNC
    __builtin_amdgcn_global_load_async_to_lds_b128(
        (__attribute__((address_space(1))) v4i_vs*)(g + (long)r * gstride + d0),
        (__attribute__((address_space(3))) v4i_vs*)(l + r * AK + d0), 0, 0);
#else
    *(uint4*)(l + r * AK + d0) = *(const uint4*)(g + (long)r * gstride + d0);
#endif
  }
}

__device__ __forceinline__ void wait_async_done() {
#if HAVE_ASYNC
#if __has_builtin(__builtin_amdgcn_s_wait_asynccnt)
  __builtin_amdgcn_s_wait_asynccnt(0);
#else
  asm volatile("s_wait_asynccnt 0x0" ::: "memory");
#endif
#endif
}

// ---------------------------------------------------------------------------
// Kernel 2: flash attention.  One block per (b*h, 64-query tile); 4 waves,
// each wave owns 16 query rows.  Online softmax, P routed via per-wave LDS.
// ---------------------------------------------------------------------------
__global__ __launch_bounds__(128) void attn_kernel(
    const __bf16* __restrict__ Qw, const __bf16* __restrict__ Kw,
    const __bf16* __restrict__ Vw, __bf16* __restrict__ AO) {
  __shared__ __bf16 Qs[64 * AK];
  __shared__ __bf16 Ks[64 * AK];
  __shared__ __bf16 Vt[64 * AK];     // [d][key]
  __shared__ __bf16 Ps[4][16 * AK];  // per-wave P tile
  const int tid = threadIdx.x;
  const int lane = tid & 31;
  const int w = tid >> 5;
  const int half = lane >> 4, ln = lane & 15;
  const int bh = blockIdx.y;  // b*H + h
  const int qt = blockIdx.x;  // 64-row query tile
  const long baseQK = (long)bh * N_ * HD_;
  const __bf16* Qg = Qw + baseQK + (long)qt * 64 * HD_;
  const __bf16* Vb = Vw + (long)bh * HD_ * N_;  // [Hd][N]

  stage_tile(Qg, HD_, Qs, tid);

  float mrun[8], lrun[8];
#pragma unroll
  for (int r = 0; r < 8; ++r) { mrun[r] = -3.0e38f; lrun[r] = 0.f; }
  v8f o[4] = {};

  for (int kt = 0; kt < 16; ++kt) {
    __syncthreads();  // previous iteration's LDS reads complete
    stage_tile(Kw + baseQK + (long)kt * 64 * HD_, HD_, Ks, tid);
    stage_tile(Vb + kt * 64, N_, Vt, tid);  // rows = d, contiguous along keys
    wait_async_done();
    __syncthreads();

    // S = Q * K^T   (Q pre-scaled by 1/sqrt(Hd))
    v16bf qa0 = frag_a(Qs + (w * 16) * AK, AK, 0);
    v16bf qa1 = frag_a(Qs + (w * 16) * AK, AK, 32);
    v8f s[4] = {};
#pragma unroll
    for (int t = 0; t < 4; ++t) {
      v16bf kb0 = frag_b(Ks + (16 * t) * AK, AK, 0);
      v16bf kb1 = frag_b(Ks + (16 * t) * AK, AK, 32);
      s[t] = wmma_bf16(qa0, kb0, s[t]);
      s[t] = wmma_bf16(qa1, kb1, s[t]);
    }

    // online softmax: rows live across 16-lane halves (C layout)
#pragma unroll
    for (int r = 0; r < 8; ++r) {
      float rm = fmaxf(fmaxf(s[0][r], s[1][r]), fmaxf(s[2][r], s[3][r]));
      rm = fmaxf(rm, __shfl_xor(rm, 1, 32));
      rm = fmaxf(rm, __shfl_xor(rm, 2, 32));
      rm = fmaxf(rm, __shfl_xor(rm, 4, 32));
      rm = fmaxf(rm, __shfl_xor(rm, 8, 32));
      float mnew = fmaxf(mrun[r], rm);
      float corr = __expf(mrun[r] - mnew);
      float rs = 0.f;
#pragma unroll
      for (int t = 0; t < 4; ++t) {
        float p = __expf(s[t][r] - mnew);
        s[t][r] = p;
        rs += p;
      }
      rs += __shfl_xor(rs, 1, 32);
      rs += __shfl_xor(rs, 2, 32);
      rs += __shfl_xor(rs, 4, 32);
      rs += __shfl_xor(rs, 8, 32);
      lrun[r] = lrun[r] * corr + rs;
      mrun[r] = mnew;
#pragma unroll
      for (int t = 0; t < 4; ++t) o[t][r] *= corr;
    }

    // reshape P: D-layout regs -> per-wave LDS -> A fragments
#pragma unroll
    for (int t = 0; t < 4; ++t)
#pragma unroll
      for (int r = 0; r < 8; ++r)
        Ps[w][(r + 8 * half) * AK + 16 * t + ln] = (__bf16)s[t][r];
    asm volatile("s_wait_dscnt 0" ::: "memory");

    v16bf pa0 = frag_a(Ps[w], AK, 0);
    v16bf pa1 = frag_a(Ps[w], AK, 32);
#pragma unroll
    for (int t = 0; t < 4; ++t) {
      v16bf vb0 = frag_b(Vt + (16 * t) * AK, AK, 0);
      v16bf vb1 = frag_b(Vt + (16 * t) * AK, AK, 32);
      o[t] = wmma_bf16(pa0, vb0, o[t]);
      o[t] = wmma_bf16(pa1, vb1, o[t]);
    }
  }

  // normalize and write attention output [B, N, D] bf16
  const int b = bh / H_, h = bh - b * H_;
#pragma unroll
  for (int r = 0; r < 8; ++r) {
    float inv = 1.0f / lrun[r];
    int nq = qt * 64 + w * 16 + r + 8 * half;
#pragma unroll
    for (int t = 0; t < 4; ++t) {
      int d = 16 * t + ln;
      AO[((long)b * N_ + nq) * D_ + h * HD_ + d] = (__bf16)(o[t][r] * inv);
    }
  }
}

// ---------------------------------------------------------------------------
// Kernel 3: output projection.  out[16384,768] = AO_bf16 @ W_proj + b_proj (fp32)
// ---------------------------------------------------------------------------
__global__ __launch_bounds__(256) void proj_kernel(
    const __bf16* __restrict__ A, const float* __restrict__ W,
    const float* __restrict__ bias, float* __restrict__ out) {
  __shared__ __bf16 As[128 * GK];
  __shared__ __bf16 Bs[128 * GK];
  const int tid = threadIdx.x;
  const int lane = tid & 31;
  const int w = tid >> 5;
  const int bm = blockIdx.y * 128;
  const int bn = blockIdx.x * 128;
  const int waveM = (w >> 1) * 32;
  const int waveN = (w & 1) * 64;

  const int ar = tid >> 2, ac8 = (tid & 3) * 8;    // A stage: uint4 (8 bf16)
  const int bk = tid >> 5, bn4 = (tid & 31) * 4;   // B stage: float4

  v8f acc[2][4] = {};

  for (int k0 = 0; k0 < D_; k0 += 32) {
    __syncthreads();
#pragma unroll
    for (int p = 0; p < 2; ++p) {
      int r = ar + 64 * p;
      *(uint4*)&As[r * GK + ac8] = *(const uint4*)&A[(long)(bm + r) * D_ + k0 + ac8];
    }
#pragma unroll
    for (int p = 0; p < 4; ++p) {
      int k = bk + 8 * p;
      const float4 v = *(const float4*)&W[(long)(k0 + k) * D_ + bn + bn4];
      Bs[(bn4 + 0) * GK + k] = (__bf16)v.x;
      Bs[(bn4 + 1) * GK + k] = (__bf16)v.y;
      Bs[(bn4 + 2) * GK + k] = (__bf16)v.z;
      Bs[(bn4 + 3) * GK + k] = (__bf16)v.w;
    }
    if (k0 + 32 < D_) {
      __builtin_prefetch(&A[(long)(bm + ar) * D_ + k0 + 32 + ac8]);
      __builtin_prefetch(&W[(long)(k0 + 32 + bk) * D_ + bn + bn4]);
    }
    __syncthreads();
    v16bf a0 = frag_a(As + (waveM + 0) * GK, GK, 0);
    v16bf a1 = frag_a(As + (waveM + 16) * GK, GK, 0);
#pragma unroll
    for (int j = 0; j < 4; ++j) {
      v16bf b = frag_b(Bs + (waveN + 16 * j) * GK, GK, 0);
      acc[0][j] = wmma_bf16(a0, b, acc[0][j]);
      acc[1][j] = wmma_bf16(a1, b, acc[1][j]);
    }
  }

  const int half = lane >> 4, ln = lane & 15;
#pragma unroll
  for (int i = 0; i < 2; ++i)
#pragma unroll
    for (int j = 0; j < 4; ++j)
#pragma unroll
      for (int r = 0; r < 8; ++r) {
        int m = bm + waveM + 16 * i + r + 8 * half;
        int c = bn + waveN + 16 * j + ln;
        out[(long)m * D_ + c] = acc[i][j][r] + bias[c];
      }
}

// ---------------------------------------------------------------------------
extern "C" void kernel_launch(void* const* d_in, const int* in_sizes, int n_in,
                              void* d_out, int out_size, void* d_ws, size_t ws_size,
                              hipStream_t stream) {
  const float* x = (const float*)d_in[0];
  const float* W_qkv = (const float*)d_in[1];
  const float* b_qkv = (const float*)d_in[2];
  const float* W_proj = (const float*)d_in[3];
  const float* b_proj = (const float*)d_in[4];
  float* out = (float*)d_out;

  const size_t qkvElems = (size_t)B_ * H_ * N_ * HD_;  // 12,582,912
  __bf16* Qw = (__bf16*)d_ws;
  __bf16* Kw = Qw + qkvElems;
  __bf16* Vw = Kw + qkvElems;   // [B,H,Hd,N]
  __bf16* AO = Vw + qkvElems;   // total ws use: ~100.7 MB

  qkv_kernel<<<dim3(18, 128), 256, 0, stream>>>(x, W_qkv, b_qkv, Qw, Kw, Vw);
  attn_kernel<<<dim3(16, 192), 128, 0, stream>>>(Qw, Kw, Vw, AO);
  proj_kernel<<<dim3(6, 128), 256, 0, stream>>>(AO, W_proj, b_proj, out);
}
